// HyperFlowNet_17875653886463
// MI455X (gfx1250) — matrix operand order repaired and checked by reading, not compiled, not run on hardware
//
#include <hip/hip_runtime.h>
#include <hip/hip_bf16.h>
#include <math.h>

// ---------------- problem constants (match reference) ----------------
#define Bn     4
#define Nn     20000
#define En     320000
#define CINn   4
#define SDn    2
#define CFn    8
#define TFn    4
#define Wn     256
#define Ln     4
#define Sn     64
#define Hn     8
#define FFNn   1024
#define OUTn   4
#define Dn     (Wn / Hn)               // 32
#define FEATn  (CINn + 2*CFn + 2*TFn)  // 28
#define BN     (Bn * Nn)               // 80000
#define EPSf   1e-6f
#define LNEPSf 1e-5f
#define NSPLIT 25                      // N/NSPLIT = 800, multiple of 32

typedef __bf16 bf16_t;
typedef __attribute__((ext_vector_type(16))) __bf16 v16bf;
typedef __attribute__((ext_vector_type(8)))  __bf16 v8bf;   // 16 bytes
typedef __attribute__((ext_vector_type(8)))  float  v8f;

union frag16 { v16bf v; struct { v8bf lo, hi; } h; };

// ---------------- WMMA fragment helpers ----------------
// A: 16x32 (MxK) bf16 from row-major src (src at tile origin; ld, origin 16B aligned).
// Lane layout: halves 0..15 = K kb..kb+15 (contiguous, matches packed B),
// kb = 16*(lane>>4). Two b128 loads.
__device__ inline v16bf load_a_frag(const bf16_t* src, int ld, int lane) {
  int m  = lane & 15;
  int kb = (lane >> 4) << 4;
  const bf16_t* p = src + (size_t)m * ld + kb;
  frag16 f;
  f.h.lo = *(const v8bf*)(p);
  f.h.hi = *(const v8bf*)(p + 8);
  return f.v;
}

// B: 32x16 fragment from PACKED layout: base[(tile*32 + lane)*16 + i], i contiguous.
__device__ inline v16bf load_b_packed(const bf16_t* base, size_t tile, int lane) {
  const bf16_t* p = base + (tile * 32 + lane) * 16;
  frag16 f;
  f.h.lo = *(const v8bf*)(p);
  f.h.hi = *(const v8bf*)(p + 8);
  return f.v;
}

__device__ inline v8f wmma_bf16(v16bf a, v16bf b, v8f c) {
  return __builtin_amdgcn_wmma_f32_16x16x32_bf16(false, a, false, b, (short)0, c,
                                                 false, false);
}

__device__ inline void zero_acc4(v8f* acc) {
#pragma unroll
  for (int j = 0; j < 4; ++j)
#pragma unroll
    for (int r = 0; r < 8; ++r) acc[j][r] = 0.0f;
}

__device__ inline float wave_sum(float s) {
#pragma unroll
  for (int off = 16; off; off >>= 1) s += __shfl_xor(s, off, 32);
  return s;
}
__device__ inline float wave_max(float s) {
#pragma unroll
  for (int off = 16; off; off >>= 1) s = fmaxf(s, __shfl_xor(s, off, 32));
  return s;
}

// ---------------- pack fp32 (KxN row-major) -> bf16 B-fragment layout ----------------
// dst[((k>>5)*(N/16) + (n>>4))*512 + ((n&15) + 16*((k&31)>>4))*16 + (k&15)]
__global__ void k_pack_b(const float* __restrict__ src, bf16_t* __restrict__ dst,
                         long K, int N) {
  long idx = (long)blockIdx.x * 256 + threadIdx.x;
  if (idx >= K * (long)N) return;
  long k = idx / N;
  int  n = (int)(idx - k * N);
  int  kr = (int)(k & 31);
  int  lane = (n & 15) + ((kr >> 4) << 4);
  int  i = kr & 15;
  size_t tile = (size_t)(k >> 5) * (N >> 4) + (n >> 4);
  dst[(tile * 32 + lane) * 16 + i] = (bf16_t)src[idx];
}

// ---------------- input embedding ----------------
__global__ void k_embed(const float* __restrict__ fields, const float* __restrict__ coords,
                        const float* __restrict__ t_norm, const float* __restrict__ freq,
                        const float* __restrict__ omega, const float* __restrict__ in_w,
                        const float* __restrict__ in_b, float* __restrict__ h) {
  int row = blockIdx.x;
  int b   = row / Nn;
  int c   = threadIdx.x;
  __shared__ float feat[FEATn];
  if (c < FEATn) {
    float f;
    if (c < CINn) {
      f = fields[(size_t)row * CINn + c];
    } else if (c < CINn + 2 * CFn) {
      int j  = c - CINn;
      int jj = j & (CFn - 1);
      float sp = 0.f;
      for (int d = 0; d < SDn; ++d) sp += coords[(size_t)row * SDn + d] * freq[d * CFn + jj];
      sp *= 6.2831853071795864769f;
      f = (j < CFn) ? sinf(sp) : cosf(sp);
    } else {
      int j  = c - CINn - 2 * CFn;
      int jj = j & (TFn - 1);
      float ang = t_norm[b] * 1000.0f * omega[jj];
      f = (j < TFn) ? sinf(ang) : cosf(ang);
    }
    feat[c] = f;
  }
  __syncthreads();
  float acc = in_b[c];
#pragma unroll
  for (int f = 0; f < FEATn; ++f) acc += feat[f] * in_w[f * Wn + c];
  h[(size_t)row * Wn + c] = acc;
}

// ---------------- LN1 + slice softmax; writes hn in PACKED-B layout ----------------
__global__ void k_ln1_slice(const float* __restrict__ h, const float* __restrict__ lnw,
                            const float* __restrict__ lnb, const float* __restrict__ slw,
                            const float* __restrict__ slb, bf16_t* __restrict__ hnB,
                            float* __restrict__ w_f32, bf16_t* __restrict__ w_bf) {
  int wave = threadIdx.x >> 5, lane = threadIdx.x & 31;
  size_t row = (size_t)blockIdx.x * 8 + wave;
  const float* hr = h + row * Wn;
  float x[8];
  float s = 0.f;
#pragma unroll
  for (int j = 0; j < 8; ++j) { x[j] = hr[lane + 32 * j]; s += x[j]; }
  float mu = wave_sum(s) * (1.0f / Wn);
  float v = 0.f;
#pragma unroll
  for (int j = 0; j < 8; ++j) { float d = x[j] - mu; v += d * d; }
  float rs = rsqrtf(wave_sum(v) * (1.0f / Wn) + LNEPSf);
  __shared__ float sh[8][Wn];
  size_t kt = row >> 5;
  int    kr = (int)(row & 31);
  int    lh = (kr >> 4) << 4;
  int    ii = kr & 15;
#pragma unroll
  for (int j = 0; j < 8; ++j) {
    int c = lane + 32 * j;
    float y = (x[j] - mu) * rs * lnw[c] + lnb[c];
    sh[wave][c] = y;
    // packed-B write: tile = kt*(Wn/16) + (c>>4); lane' = (c&15)+lh; elem ii
    hnB[(((kt << 4) + (c >> 4)) * 32 + (c & 15) + lh) * 16 + ii] = (bf16_t)y;
  }
  __syncthreads();
  float l0 = slb[lane], l1 = slb[lane + 32];
  for (int k = 0; k < Wn; ++k) {
    float hv = sh[wave][k];
    l0 += hv * slw[k * Sn + lane];
    l1 += hv * slw[k * Sn + lane + 32];
  }
  float m  = wave_max(fmaxf(l0, l1));
  float e0 = expf(l0 - m), e1 = expf(l1 - m);
  float inv = 1.0f / wave_sum(e0 + e1);
  float w0 = e0 * inv, w1 = e1 * inv;
  w_f32[row * Sn + lane]      = w0;
  w_f32[row * Sn + lane + 32] = w1;
  w_bf [row * Sn + lane]      = (bf16_t)w0;
  w_bf [row * Sn + lane + 32] = (bf16_t)w1;
}

// ---------------- LN2 (row-major bf16 out, feeds FFN LDS tile) ----------------
__global__ void k_ln2(const float* __restrict__ h, const float* __restrict__ lnw,
                      const float* __restrict__ lnb, bf16_t* __restrict__ hn_bf) {
  int wave = threadIdx.x >> 5, lane = threadIdx.x & 31;
  size_t row = (size_t)blockIdx.x * 8 + wave;
  const float* hr = h + row * Wn;
  float x[8];
  float s = 0.f;
#pragma unroll
  for (int j = 0; j < 8; ++j) { x[j] = hr[lane + 32 * j]; s += x[j]; }
  float mu = wave_sum(s) * (1.0f / Wn);
  float v = 0.f;
#pragma unroll
  for (int j = 0; j < 8; ++j) { float d = x[j] - mu; v += d * d; }
  float rs = rsqrtf(wave_sum(v) * (1.0f / Wn) + LNEPSf);
#pragma unroll
  for (int j = 0; j < 8; ++j) {
    int c = lane + 32 * j;
    hn_bf[row * Wn + c] = (bf16_t)((x[j] - mu) * rs * lnw[c] + lnb[c]);
  }
}

// ---------------- transpose weights: wT[b,s,n] = w[b,n,s] (bf16) ----------------
__global__ void k_transpose_w(const float* __restrict__ w_f32, bf16_t* __restrict__ wT) {
  // grid (Nn/32, Bn), block 256; tile 32(n) x 64(s)
  int n0 = blockIdx.x * 32, b = blockIdx.y, t = threadIdx.x;
  __shared__ float sh[32][Sn + 1];
  const float* src = w_f32 + ((size_t)b * Nn + n0) * Sn;
#pragma unroll
  for (int p = 0; p < 8; ++p) {
    int idx = t + p * 256;
    int nl = idx >> 6, s = idx & 63;
    sh[nl][s] = src[(size_t)nl * Sn + s];
  }
  __syncthreads();
#pragma unroll
  for (int p = 0; p < 8; ++p) {
    int idx = t + p * 256;
    int s = idx >> 5, nl = idx & 31;
    wT[((size_t)b * Sn + s) * Nn + n0 + nl] = (bf16_t)sh[nl][s];
  }
}

// ------- out[s,c] += sum_n wT[s,n] * Xpacked[n,c]  (WMMA, split-K atomics) -------
__global__ void k_at_b(const bf16_t* __restrict__ wT, const bf16_t* __restrict__ Xp,
                       float* __restrict__ out, int C) {
  int lane = threadIdx.x;
  int s0 = blockIdx.x * 16, c0 = blockIdx.y * 64;
  int b = blockIdx.z / NSPLIT, chunk = blockIdx.z % NSPLIT;
  const int npc = Nn / NSPLIT;  // 800
  const bf16_t* ap = wT + ((size_t)b * Sn + s0) * Nn;
  const bf16_t* xp = Xp + (size_t)b * Nn * C;
  int ntiles = C >> 4;
  v8f acc[4]; zero_acc4(acc);
  int k0 = chunk * npc;
  for (int k = k0; k < k0 + npc; k += 32) {
    v16bf a = load_a_frag(ap + k, Nn, lane);
    size_t tbase = (size_t)(k >> 5) * ntiles + (c0 >> 4);
#pragma unroll
    for (int j = 0; j < 4; ++j)
      acc[j] = wmma_bf16(a, load_b_packed(xp, tbase + j, lane), acc[j]);
  }
  int n = lane & 15, mb = (lane >> 4) * 8;
  float* op = out + (size_t)b * Sn * C;
#pragma unroll
  for (int j = 0; j < 4; ++j)
#pragma unroll
    for (int r = 0; r < 8; ++r)
      atomicAdd(&op[(size_t)(s0 + mb + r) * C + c0 + 16 * j + n], acc[j][r]);
}

// ---------------- wsum[b,s] = sum_n w[b,n,s] ----------------
__global__ void k_wsum(const float* __restrict__ w_f32, float* __restrict__ wsum) {
  int b = blockIdx.x, chunk = blockIdx.y, s = threadIdx.x;
  const int npc = Nn / NSPLIT;
  const float* wp = w_f32 + ((size_t)b * Nn + (size_t)chunk * npc) * Sn + s;
  float acc = 0.f;
  for (int n = 0; n < npc; ++n) acc += wp[(size_t)n * Sn];
  atomicAdd(&wsum[b * Sn + s], acc);
}

// ---------------- slices_bf = slices_f32 / max(wsum, eps) (row-major) ----------------
__global__ void k_fin_slices(const float* __restrict__ sl, const float* __restrict__ wsum,
                             bf16_t* __restrict__ sl_bf) {
  int i = blockIdx.x * blockDim.x + threadIdx.x;
  if (i >= Bn * Sn * Wn) return;
  int b = i / (Sn * Wn);
  int s = (i / Wn) % Sn;
  sl_bf[i] = (bf16_t)(sl[i] / fmaxf(wsum[b * Sn + s], EPSf));
}

// ---------------- sparse aggregate ----------------
__global__ void k_gw(const int* __restrict__ adj, const float* __restrict__ vals,
                     const float* __restrict__ w_f32, float* __restrict__ gw) {
  long t = (long)blockIdx.x * 256 + threadIdx.x;
  int e = (int)(t >> 6), s = (int)(t & 63);
  if (e >= En) return;
  int r = adj[e], c = adj[En + e];
  float vv = vals[e];
#pragma unroll
  for (int b = 0; b < Bn; ++b)
    atomicAdd(&gw[((size_t)b * Nn + r) * Sn + s],
              vv * w_f32[((size_t)b * Nn + c) * Sn + s]);
}

// ---------------- gb finalize ----------------
__global__ void k_gb_fin(const float* __restrict__ gb, float* __restrict__ gb2) {
  int b = blockIdx.x >> 6, s = blockIdx.x & 63, t = threadIdx.x;
  const float* g = gb + (size_t)b * Sn * Sn;
  float v = 0.5f * (g[s * Sn + t] + g[t * Sn + s]);
  __shared__ float sh[Sn];
  sh[t] = v;
  __syncthreads();
  for (int off = 32; off; off >>= 1) {
    if (t < off) sh[t] += sh[t + off];
    __syncthreads();
  }
  float norm = fmaxf(sh[0], EPSf);
  gb2[(size_t)b * Sn * Sn + s * Sn + t] = logf(fmaxf(v / norm, EPSf));
}

// ------- generic small GEMM: out = A(MxK)@Bpacked(KxN)+bias; row-major or packed out -------
__global__ void k_gemm(const bf16_t* __restrict__ A, int lda, long strideA,
                       const bf16_t* __restrict__ Bp, long strideB,
                       const float* __restrict__ bias,
                       bf16_t* __restrict__ out, int No, long strideO, int K,
                       int packed_out) {
  int lane = threadIdx.x;
  int m0 = blockIdx.x * 16, n0 = blockIdx.y * 64, b = blockIdx.z;
  const bf16_t* Ap = A + (size_t)b * strideA + (size_t)m0 * lda;
  const bf16_t* Bpp = Bp + (size_t)b * strideB;
  int ntiles = No >> 4;
  v8f acc[4]; zero_acc4(acc);
  for (int k = 0; k < K; k += 32) {
    v16bf a = load_a_frag(Ap + k, lda, lane);
    size_t tbase = (size_t)(k >> 5) * ntiles + (n0 >> 4);
#pragma unroll
    for (int j = 0; j < 4; ++j)
      acc[j] = wmma_bf16(a, load_b_packed(Bpp, tbase + j, lane), acc[j]);
  }
  int n = lane & 15, mb = (lane >> 4) * 8;
  bf16_t* op = out + (size_t)b * strideO;
#pragma unroll
  for (int j = 0; j < 4; ++j)
#pragma unroll
    for (int r = 0; r < 8; ++r) {
      int nn = n0 + 16 * j + n;
      float val = acc[j][r] + bias[nn];
      if (packed_out) {
        int kk = m0 + mb + r;
        int lane2 = (nn & 15) + (((kk & 31) >> 4) << 4);
        op[(((size_t)(kk >> 5) * ntiles + (nn >> 4)) * 32 + lane2) * 16 + (kk & 15)] =
            (bf16_t)val;
      } else {
        op[(size_t)(m0 + mb + r) * No + nn] = (bf16_t)val;
      }
    }
}

// ---------------- attention over S=64 slices (tiny; VALU) ----------------
__global__ void k_attn(const bf16_t* __restrict__ q, const bf16_t* __restrict__ k,
                       const bf16_t* __restrict__ v, const float* __restrict__ gb2,
                       const float* __restrict__ beta_raw, int l, bf16_t* __restrict__ so) {
  int bid = blockIdx.x;
  int s  = bid & 63;
  int hh = (bid >> 6) & 7;
  int b  = bid >> 9;
  int t  = threadIdx.x;
  const bf16_t* qp = q + ((size_t)b * Sn + s) * Wn + hh * Dn;
  const bf16_t* kp = k + ((size_t)b * Sn + t) * Wn + hh * Dn;
  float acc = 0.f;
#pragma unroll
  for (int d = 0; d < Dn; ++d) acc += (float)qp[d] * (float)kp[d];
  float beta  = log1pf(expf(beta_raw[l]));
  float logit = acc * 0.17677669529663687f
              + beta * gb2[((size_t)b * Sn + s) * Sn + t];
  __shared__ float sh[Sn];
  sh[t] = logit; __syncthreads();
  float m = -3.0e38f;
  for (int i = 0; i < Sn; ++i) m = fmaxf(m, sh[i]);
  float e = expf(logit - m);
  __syncthreads(); sh[t] = e; __syncthreads();
  float sum = 0.f;
  for (int i = 0; i < Sn; ++i) sum += sh[i];
  float a = e / sum;
  __syncthreads(); sh[t] = a; __syncthreads();
  if (t < Dn) {
    float o = 0.f;
    for (int tt = 0; tt < Sn; ++tt)
      o += sh[tt] * (float)v[((size_t)b * Sn + tt) * Wn + hh * Dn + t];
    so[((size_t)b * Sn + s) * Wn + hh * Dn + t] = (bf16_t)o;
  }
}

// -------- de-slice: h[b,n,:] += w[b,n,:](A, row-major ld=64) @ so2packed(64x256) --------
__global__ void k_deslice(const bf16_t* __restrict__ wts, const bf16_t* __restrict__ so2p,
                          float* __restrict__ h) {
  int wave = threadIdx.x >> 5, lane = threadIdx.x & 31;
  size_t r0 = (size_t)blockIdx.x * 16;
  int b = (int)(r0 / Nn);
  const bf16_t* Ap = wts + r0 * Sn;
  const bf16_t* Bp = so2p + (size_t)b * Sn * Wn;
  v8f acc[4]; zero_acc4(acc);
  for (int k = 0; k < Sn; k += 32) {
    v16bf a = load_a_frag(Ap + k, Sn, lane);
    size_t tbase = (size_t)(k >> 5) * (Wn >> 4) + ((wave * 64) >> 4);
#pragma unroll
    for (int j = 0; j < 4; ++j)
      acc[j] = wmma_bf16(a, load_b_packed(Bp, tbase + j, lane), acc[j]);
  }
  int n = lane & 15, mb = (lane >> 4) * 8;
#pragma unroll
  for (int j = 0; j < 4; ++j)
#pragma unroll
    for (int r = 0; r < 8; ++r) {
      size_t idx = (r0 + mb + r) * Wn + wave * 64 + 16 * j + n;
      h[idx] += acc[j][r];
    }
}

// --------- fused FFN: h += W2 @ gelu(W1 @ hn2 + b1) + b2  (WMMA, LDS staged) ---------
__global__ void k_ffn(const bf16_t* __restrict__ hn2, const bf16_t* __restrict__ w1p,
                      const float* __restrict__ b1, const bf16_t* __restrict__ w2p,
                      const float* __restrict__ b2, float* __restrict__ h) {
  __shared__ __align__(16) bf16_t a_tile[16 * Wn];     // 8 KB
  __shared__ __align__(16) bf16_t g_tile[16 * FFNn];   // 32 KB
  int tid = threadIdx.x, wave = tid >> 5, lane = tid & 31;
  size_t r0 = (size_t)blockIdx.x * 16;
  {
    const uint4* src = (const uint4*)(hn2 + r0 * Wn);
    uint4* dst = (uint4*)a_tile;
    for (int i = tid; i < (16 * Wn) / 8; i += 128) dst[i] = src[i];
  }
  __syncthreads();
  // phase 1: wave computes intermediate cols [256*wave, 256*wave+256)
  for (int cc = 0; cc < 4; ++cc) {
    int n0 = wave * 256 + cc * 64;
    v8f acc[4]; zero_acc4(acc);
    for (int k = 0; k < Wn; k += 32) {
      v16bf a = load_a_frag(a_tile + k, Wn, lane);
      size_t tbase = (size_t)(k >> 5) * (FFNn >> 4) + (n0 >> 4);
#pragma unroll
      for (int j = 0; j < 4; ++j)
        acc[j] = wmma_bf16(a, load_b_packed(w1p, tbase + j, lane), acc[j]);
    }
    int n = lane & 15, mb = (lane >> 4) * 8;
#pragma unroll
    for (int j = 0; j < 4; ++j)
#pragma unroll
      for (int r = 0; r < 8; ++r) {
        int col = n0 + 16 * j + n;
        float x = acc[j][r] + b1[col];
        float g = 0.5f * x * (1.0f + erff(x * 0.70710678118654752f));
        g_tile[(mb + r) * FFNn + col] = (bf16_t)g;
      }
  }
  __syncthreads();
  // phase 2: wave computes output cols [64*wave, 64*wave+64), K = 1024
  {
    int n0 = wave * 64;
    v8f acc[4]; zero_acc4(acc);
    for (int k = 0; k < FFNn; k += 32) {
      v16bf a = load_a_frag(g_tile + k, FFNn, lane);
      size_t tbase = (size_t)(k >> 5) * (Wn >> 4) + (n0 >> 4);
#pragma unroll
      for (int j = 0; j < 4; ++j)
        acc[j] = wmma_bf16(a, load_b_packed(w2p, tbase + j, lane), acc[j]);
    }
    int n = lane & 15, mb = (lane >> 4) * 8;
#pragma unroll
    for (int j = 0; j < 4; ++j)
#pragma unroll
      for (int r = 0; r < 8; ++r) {
        int col = n0 + 16 * j + n;
        size_t idx = (r0 + mb + r) * Wn + col;
        h[idx] += acc[j][r] + b2[col];
      }
  }
}

// ---------------- final projection ----------------
__global__ void k_out(const float* __restrict__ h, const float* __restrict__ ow,
                      const float* __restrict__ ob, float* __restrict__ out) {
  int wave = threadIdx.x >> 5, lane = threadIdx.x & 31;
  size_t row = (size_t)blockIdx.x * 8 + wave;
  const float* hr = h + row * Wn;
  float a0 = 0.f, a1 = 0.f, a2 = 0.f, a3 = 0.f;
#pragma unroll
  for (int j = 0; j < 8; ++j) {
    int c = lane + 32 * j;
    float hv = hr[c];
    a0 += hv * ow[c * OUTn + 0];
    a1 += hv * ow[c * OUTn + 1];
    a2 += hv * ow[c * OUTn + 2];
    a3 += hv * ow[c * OUTn + 3];
  }
  a0 = wave_sum(a0); a1 = wave_sum(a1); a2 = wave_sum(a2); a3 = wave_sum(a3);
  if (lane == 0) {
    float* o = out + row * OUTn;
    o[0] = a0 + ob[0]; o[1] = a1 + ob[1]; o[2] = a2 + ob[2]; o[3] = a3 + ob[3];
  }
}

// =================================================================================
extern "C" void kernel_launch(void* const* d_in, const int* in_sizes, int n_in,
                              void* d_out, int out_size, void* d_ws, size_t ws_size,
                              hipStream_t stream) {
  const float* fields     = (const float*)d_in[0];
  const float* coords     = (const float*)d_in[1];
  const float* t_norm     = (const float*)d_in[2];
  const float* adj_values = (const float*)d_in[3];
  const float* freq       = (const float*)d_in[4];
  const float* omega      = (const float*)d_in[5];
  const float* in_w       = (const float*)d_in[6];
  const float* in_b       = (const float*)d_in[7];
  const float* ln1_w      = (const float*)d_in[8];
  const float* ln1_b      = (const float*)d_in[9];
  const float* ln2_w      = (const float*)d_in[10];
  const float* ln2_b      = (const float*)d_in[11];
  const float* slice_w    = (const float*)d_in[12];
  const float* slice_b    = (const float*)d_in[13];
  const float* q_w        = (const float*)d_in[14];
  const float* q_b        = (const float*)d_in[15];
  const float* k_w        = (const float*)d_in[16];
  const float* k_b        = (const float*)d_in[17];
  const float* v_w        = (const float*)d_in[18];
  const float* v_b        = (const float*)d_in[19];
  const float* o_w        = (const float*)d_in[20];
  const float* o_b        = (const float*)d_in[21];
  const float* beta_raw   = (const float*)d_in[22];
  const float* ffn1_w     = (const float*)d_in[23];
  const float* ffn1_b     = (const float*)d_in[24];
  const float* ffn2_w     = (const float*)d_in[25];
  const float* ffn2_b     = (const float*)d_in[26];
  const float* out_w      = (const float*)d_in[27];
  const float* out_b      = (const float*)d_in[28];
  const int*   adj_idx    = (const int*)d_in[29];

  // -------- workspace layout --------
  char* base = (char*)d_ws;
  size_t off = 0;
  auto alloc = [&](size_t bytes) -> char* {
    char* p = base + off;
    off = (off + bytes + 255) & ~(size_t)255;
    return p;
  };
  float*  h_f    = (float*) alloc((size_t)BN * Wn * 4);
  bf16_t* hn_ws  = (bf16_t*)alloc((size_t)BN * Wn * 2);  // packed-B (ln1) / row-major (ln2)
  float*  w_f    = (float*) alloc((size_t)BN * Sn * 4);
  bf16_t* w_bf   = (bf16_t*)alloc((size_t)BN * Sn * 2);  // row-major (deslice A)
  bf16_t* wT_bf  = (bf16_t*)alloc((size_t)Bn * Sn * Nn * 2);  // (B,S,N) row-major
  float*  gw_f   = (float*) alloc((size_t)BN * Sn * 4);
  bf16_t* gwB    = (bf16_t*)alloc((size_t)BN * Sn * 2);  // packed-B
  float*  sl_f   = (float*) alloc((size_t)Bn * Sn * Wn * 4);
  bf16_t* sl_bf  = (bf16_t*)alloc((size_t)Bn * Sn * Wn * 2);
  float*  wsum   = (float*) alloc((size_t)Bn * Sn * 4);
  float*  gb_raw = (float*) alloc((size_t)Bn * Sn * Sn * 4);
  float*  gb2    = (float*) alloc((size_t)Bn * Sn * Sn * 4);
  bf16_t* q_bf   = (bf16_t*)alloc((size_t)Bn * Sn * Wn * 2);
  bf16_t* kk_bf  = (bf16_t*)alloc((size_t)Bn * Sn * Wn * 2);
  bf16_t* v_bf   = (bf16_t*)alloc((size_t)Bn * Sn * Wn * 2);
  bf16_t* soA_bf = (bf16_t*)alloc((size_t)Bn * Sn * Wn * 2);
  bf16_t* so2p   = (bf16_t*)alloc((size_t)Bn * Sn * Wn * 2);  // packed-B
  bf16_t* qw_p   = (bf16_t*)alloc((size_t)Ln * Wn * Wn * 2);
  bf16_t* kw_p   = (bf16_t*)alloc((size_t)Ln * Wn * Wn * 2);
  bf16_t* vw_p   = (bf16_t*)alloc((size_t)Ln * Wn * Wn * 2);
  bf16_t* ow_p   = (bf16_t*)alloc((size_t)Ln * Wn * Wn * 2);
  bf16_t* f1w_p  = (bf16_t*)alloc((size_t)Ln * Wn * FFNn * 2);
  bf16_t* f2w_p  = (bf16_t*)alloc((size_t)Ln * FFNn * Wn * 2);
  (void)ws_size; (void)in_sizes; (void)n_in; (void)out_size;

  // -------- weight packing to bf16 B-fragment layout --------
  {
    long nww = (long)Ln * Wn * Wn, nf = (long)Ln * Wn * FFNn;
    int gww = (int)((nww + 255) / 256), gf = (int)((nf + 255) / 256);
    k_pack_b<<<gww, 256, 0, stream>>>(q_w,    qw_p,  (long)Ln * Wn,   Wn);
    k_pack_b<<<gww, 256, 0, stream>>>(k_w,    kw_p,  (long)Ln * Wn,   Wn);
    k_pack_b<<<gww, 256, 0, stream>>>(v_w,    vw_p,  (long)Ln * Wn,   Wn);
    k_pack_b<<<gww, 256, 0, stream>>>(o_w,    ow_p,  (long)Ln * Wn,   Wn);
    k_pack_b<<<gf,  256, 0, stream>>>(ffn1_w, f1w_p, (long)Ln * Wn,   FFNn);
    k_pack_b<<<gf,  256, 0, stream>>>(ffn2_w, f2w_p, (long)Ln * FFNn, Wn);
  }

  // -------- embedding --------
  k_embed<<<BN, Wn, 0, stream>>>(fields, coords, t_norm, freq, omega, in_w, in_b, h_f);

  // -------- transformer layers --------
  for (int l = 0; l < Ln; ++l) {
    k_ln1_slice<<<BN / 8, 256, 0, stream>>>(h_f, ln1_w + l * Wn, ln1_b + l * Wn,
                                            slice_w + (size_t)l * Wn * Sn, slice_b + l * Sn,
                                            hn_ws, w_f, w_bf);
    k_transpose_w<<<dim3(Nn / 32, Bn), 256, 0, stream>>>(w_f, wT_bf);
    hipMemsetAsync(sl_f,   0, (size_t)Bn * Sn * Wn * 4, stream);
    hipMemsetAsync(wsum,   0, (size_t)Bn * Sn * 4,      stream);
    hipMemsetAsync(gw_f,   0, (size_t)BN * Sn * 4,      stream);
    hipMemsetAsync(gb_raw, 0, (size_t)Bn * Sn * Sn * 4, stream);

    // slices += w^T @ hn
    k_at_b<<<dim3(Sn / 16, Wn / 64, Bn * NSPLIT), 32, 0, stream>>>(wT_bf, hn_ws, sl_f, Wn);
    k_wsum<<<dim3(Bn, NSPLIT), Sn, 0, stream>>>(w_f, wsum);
    k_fin_slices<<<(Bn * Sn * Wn + 255) / 256, 256, 0, stream>>>(sl_f, wsum, sl_bf);

    // graph bias path
    k_gw<<<(int)(((long)En * Sn) / 256), 256, 0, stream>>>(adj_idx, adj_values, w_f, gw_f);
    k_pack_b<<<(int)(((long)BN * Sn + 255) / 256), 256, 0, stream>>>(gw_f, gwB,
                                                                     (long)BN, Sn);
    k_at_b<<<dim3(Sn / 16, Sn / 64, Bn * NSPLIT), 32, 0, stream>>>(wT_bf, gwB, gb_raw, Sn);
    k_gb_fin<<<Bn * Sn, Sn, 0, stream>>>(gb_raw, gb2);

    // q/k/v projections (M=64, N=256, K=256)
    k_gemm<<<dim3(Sn / 16, Wn / 64, Bn), 32, 0, stream>>>(
        sl_bf, Wn, (long)Sn * Wn, qw_p + (size_t)l * Wn * Wn, 0,
        q_b + l * Wn, q_bf, Wn, (long)Sn * Wn, Wn, 0);
    k_gemm<<<dim3(Sn / 16, Wn / 64, Bn), 32, 0, stream>>>(
        sl_bf, Wn, (long)Sn * Wn, kw_p + (size_t)l * Wn * Wn, 0,
        k_b + l * Wn, kk_bf, Wn, (long)Sn * Wn, Wn, 0);
    k_gemm<<<dim3(Sn / 16, Wn / 64, Bn), 32, 0, stream>>>(
        sl_bf, Wn, (long)Sn * Wn, vw_p + (size_t)l * Wn * Wn, 0,
        v_b + l * Wn, v_bf, Wn, (long)Sn * Wn, Wn, 0);

    k_attn<<<Bn * Hn * Sn, Sn, 0, stream>>>(q_bf, kk_bf, v_bf, gb2, beta_raw, l, soA_bf);

    // output projection (packed-B output for de-slice)
    k_gemm<<<dim3(Sn / 16, Wn / 64, Bn), 32, 0, stream>>>(
        soA_bf, Wn, (long)Sn * Wn, ow_p + (size_t)l * Wn * Wn, 0,
        o_b + l * Wn, so2p, Wn, (long)Sn * Wn, Wn, 1);

    // h += weights @ so2
    k_deslice<<<BN / 16, 128, 0, stream>>>(w_bf, so2p, h_f);

    // FFN (fused, residual)
    k_ln2<<<BN / 8, 256, 0, stream>>>(h_f, ln2_w + l * Wn, ln2_b + l * Wn, hn_ws);
    k_ffn<<<BN / 16, 128, 0, stream>>>(hn_ws, f1w_p + (size_t)l * Wn * FFNn,
                                       ffn1_b + l * FFNn,
                                       f2w_p + (size_t)l * FFNn * Wn,
                                       ffn2_b + l * Wn, h_f);
  }

  k_out<<<BN / 8, 256, 0, stream>>>(h_f, out_w, out_b, (float*)d_out);
}
